// Transformer_11665131176468
// MI455X (gfx1250) — matrix operand order repaired and checked by reading, not compiled, not run on hardware
//
#include <hip/hip_runtime.h>
#include <hip/hip_bf16.h>

// ---------------------------------------------------------------------------
// CDNA5 (gfx1250) transformer forward: bf16 WMMA GEMMs (async-LDS double
// buffered) + WMMA flash attention.
// D = 8 layers, B=2, N=2048, DIM=1024, HEADS=16, DHEAD=64, MLP=4096.
// ---------------------------------------------------------------------------

typedef __bf16 bf16_t;
typedef __attribute__((ext_vector_type(16))) __bf16 v16bf;
typedef __attribute__((ext_vector_type(8)))  float  v8f;
typedef __attribute__((ext_vector_type(4)))  float  f32x4;

union FragU { v16bf v; f32x4 f4[2]; };

// A-fragment (16x32 bf16): lane holds k = half*8 + [0..7] and 16+half*8+[0..7].
// p must already include  row*stride + half*8  (+ k0).
__device__ __forceinline__ v16bf ld_a_frag(const bf16_t* p) {
  FragU u;
  u.f4[0] = *reinterpret_cast<const f32x4*>(p);
  u.f4[1] = *reinterpret_cast<const f32x4*>(p + 16);
  return u.v;
}

// B-fragment (32x16 bf16): lane = column, holds k = half*8 + [0..7] and 16+half*8+[0..7].
// p must already include  col*stride + half*8  (+ k0).
__device__ __forceinline__ v16bf ld_b_frag(const bf16_t* p) {
  FragU u;
  u.f4[0] = *reinterpret_cast<const f32x4*>(p);
  u.f4[1] = *reinterpret_cast<const f32x4*>(p + 16);
  return u.v;
}

__device__ __forceinline__ v8f wmma_bf16(v16bf a, v16bf b, v8f c) {
  // D = A(16x32 bf16) * B(32x16 bf16) + C(16x16 f32)
  return __builtin_amdgcn_wmma_f32_16x16x32_bf16(
      /*neg_a=*/false, a, /*neg_b=*/false, b,
      /*c_mod=*/(short)0, c, /*reuse_a=*/false, /*reuse_b=*/false);
}

// Async DMA: 16 bytes per lane, global -> LDS, tracked by ASYNCcnt.
__device__ __forceinline__ void async_ld_b128(unsigned lds_off, const char* g) {
  asm volatile("global_load_async_to_lds_b128 %0, %1, off"
               :: "v"(lds_off), "v"(g) : "memory");
}

__device__ __forceinline__ float gelu_exact(float x) {
  return 0.5f * x * (1.0f + erff(x * 0.70710678118654752440f));
}

// ---------------------------------------------------------------------------
// GEMM:  out[M,N] = A[M,K](bf16) * Bt[N,K](bf16, pre-transposed weights)
// EPI 0: store bf16
// EPI 1: +bias, exact GELU, store bf16
// EPI 2: +bias, +residual(f32), store f32
// Block: 256 thr = 8 waves (2x4), wave tile 64x32, block tile 128x128.
// A/B tiles (128x32 each) staged in LDS via global_load_async_to_lds_b128,
// double buffered: DMA of tile kt+1 overlaps the 8 WMMAs of tile kt.
// ---------------------------------------------------------------------------
template<int EPI>
__global__ __launch_bounds__(256) void gemm_bf16_kernel(
    const bf16_t* __restrict__ A, const bf16_t* __restrict__ Bt,
    const float* __restrict__ bias, const float* res,
    void* out, int M, int N, int K)
{
  __shared__ __align__(16) bf16_t As[2][128 * 32];   // 2 x 8 KB
  __shared__ __align__(16) bf16_t Bs[2][128 * 32];   // 2 x 8 KB

  const int lane = threadIdx.x & 31;
  const int wave = threadIdx.x >> 5;
  const int wm   = wave >> 2;            // 0..1
  const int wn   = wave & 3;             // 0..3
  const int bmblk = blockIdx.y * 128;
  const int bnblk = blockIdx.x * 128;
  const int half = lane >> 4;
  const int r    = lane & 15;

  // ---- DMA lane mapping: wave w stages rows [w*16, w*16+16) of A and B.
  // One b128 per lane covers 8 rows x 64B; two issues per matrix per tile.
  const int lrow = wave * 16 + (lane >> 2);     // 0..127 (first 8-row group)
  const int lcol = (lane & 3) * 16;             // byte offset within 64B row
  const char* gA = (const char*)(A  + (size_t)(bmblk + lrow) * K) + lcol;
  const char* gB = (const char*)(Bt + (size_t)(bnblk + lrow) * K) + lcol;
  const unsigned aBase =
      (unsigned)(uintptr_t)(&As[0][0]) + (unsigned)(lrow * 64 + lcol);
  const unsigned bBase =
      (unsigned)(uintptr_t)(&Bs[0][0]) + (unsigned)(lrow * 64 + lcol);
  const size_t row8 = (size_t)8 * K * 2;        // +8 rows, in bytes

  auto prefetch = [&](int buf, int k0) {
    const char* pa = gA + (size_t)k0 * 2;
    const char* pb = gB + (size_t)k0 * 2;
    const unsigned la = aBase + (unsigned)buf * 8192u;
    const unsigned lb = bBase + (unsigned)buf * 8192u;
    async_ld_b128(la,        pa);
    async_ld_b128(la + 512u, pa + row8);
    async_ld_b128(lb,        pb);
    async_ld_b128(lb + 512u, pb + row8);
  };

  // ---- compute-side LDS fragment offsets (elements)
  const int aoffs = (wm * 64 + r) * 32 + half * 8;    // + i*512
  const int boffs = (wn * 32 + r) * 32 + half * 8;    // + j*512

  v8f c[4][2] = {};
  const int nt = K >> 5;

  prefetch(0, 0);
  for (int kt = 0; kt < nt; ++kt) {
    const int buf = kt & 1;
    __syncthreads();                    // all reads of buf^1 (tile kt-1) done
    if (kt + 1 < nt) {
      prefetch(buf ^ 1, (kt + 1) << 5); // DMA next tile while computing this one
      asm volatile("s_wait_asynccnt 4" ::: "memory");  // tile kt landed
    } else {
      asm volatile("s_wait_asynccnt 0" ::: "memory");
    }
    __syncthreads();                    // tile kt visible to all waves

    const bf16_t* Ab = &As[buf][0];
    const bf16_t* Bb = &Bs[buf][0];
    v16bf a[4], b[2];
#pragma unroll
    for (int i = 0; i < 4; ++i) a[i] = ld_a_frag(Ab + aoffs + i * 512);
#pragma unroll
    for (int j = 0; j < 2; ++j) b[j] = ld_b_frag(Bb + boffs + j * 512);
#pragma unroll
    for (int i = 0; i < 4; ++i)
#pragma unroll
      for (int j = 0; j < 2; ++j)
        c[i][j] = wmma_bf16(a[i], b[j], c[i][j]);
  }

  // C layout: VGPR t -> row (bm + i*16 + half*8 + t), col (bn + j*16 + r)
  const int bm = bmblk + wm * 64;
  const int bn = bnblk + wn * 32;
#pragma unroll
  for (int j = 0; j < 2; ++j) {
    const int col = bn + j * 16 + r;
    float bs = 0.0f;
    if (EPI != 0) bs = bias[col];
#pragma unroll
    for (int i = 0; i < 4; ++i) {
#pragma unroll
      for (int t = 0; t < 8; ++t) {
        const int row = bm + i * 16 + half * 8 + t;
        float v = c[i][j][t] + bs;
        if (EPI == 1) v = gelu_exact(v);
        const size_t idx = (size_t)row * N + col;
        if (EPI == 2) {
          reinterpret_cast<float*>(out)[idx] = v + res[idx];
        } else {
          reinterpret_cast<bf16_t*>(out)[idx] = (bf16_t)v;
        }
      }
    }
  }
}

// ---------------------------------------------------------------------------
// LayerNorm over DIM=1024, fp32 in -> bf16 out. One block (256 thr) per row.
// ---------------------------------------------------------------------------
__global__ __launch_bounds__(256) void layernorm_kernel(
    const float* __restrict__ x, const float* __restrict__ g,
    const float* __restrict__ b, bf16_t* __restrict__ out)
{
  const int row = blockIdx.x;
  const float* xr = x + (size_t)row * 1024;
  const f32x4 v = reinterpret_cast<const f32x4*>(xr)[threadIdx.x];
  float s  = v.x + v.y + v.z + v.w;
  float s2 = v.x * v.x + v.y * v.y + v.z * v.z + v.w * v.w;
#pragma unroll
  for (int m = 16; m >= 1; m >>= 1) {
    s  += __shfl_xor(s, m, 32);
    s2 += __shfl_xor(s2, m, 32);
  }
  __shared__ float red[2][8];
  const int wave = threadIdx.x >> 5, lane = threadIdx.x & 31;
  if (lane == 0) { red[0][wave] = s; red[1][wave] = s2; }
  __syncthreads();
  s = 0.0f; s2 = 0.0f;
#pragma unroll
  for (int w = 0; w < 8; ++w) { s += red[0][w]; s2 += red[1][w]; }
  const float mean = s * (1.0f / 1024.0f);
  const float var  = s2 * (1.0f / 1024.0f) - mean * mean;
  const float rstd = rsqrtf(var + 1e-5f);
  const int i0 = threadIdx.x * 4;
  const float xv[4] = {v.x, v.y, v.z, v.w};
#pragma unroll
  for (int t = 0; t < 4; ++t) {
    const float y = (xv[t] - mean) * rstd * g[i0 + t] + b[i0 + t];
    out[(size_t)row * 1024 + i0 + t] = (bf16_t)y;
  }
}

// ---------------------------------------------------------------------------
// Transpose + fp32->bf16: in[K][N] -> out[N][K].  Block (32,8), LDS-tiled.
// ---------------------------------------------------------------------------
__global__ __launch_bounds__(256) void transpose_convert_kernel(
    const float* __restrict__ in, bf16_t* __restrict__ out, int K, int N)
{
  __shared__ float tile[32][33];
  const int n0 = blockIdx.x * 32;
  const int k0 = blockIdx.y * 32;
  const int tx = threadIdx.x, ty = threadIdx.y;
#pragma unroll
  for (int i = 0; i < 4; ++i)
    tile[ty + 8 * i][tx] = in[(size_t)(k0 + ty + 8 * i) * N + n0 + tx];
  __syncthreads();
#pragma unroll
  for (int i = 0; i < 4; ++i)
    out[(size_t)(n0 + ty + 8 * i) * K + k0 + tx] = (bf16_t)tile[tx][ty + 8 * i];
}

// ---------------------------------------------------------------------------
// Build V^T: vt[(b*16+h)*64 + d][n] = qkv[b*2048+n][2048 + h*64 + d]  (bf16).
// Grid: (2048/32, 64/32, B*H), block (32,8).
// ---------------------------------------------------------------------------
__global__ __launch_bounds__(256) void build_vt_kernel(
    const bf16_t* __restrict__ qkv, bf16_t* __restrict__ vt)
{
  __shared__ bf16_t tile[32][34];
  const int n0 = blockIdx.x * 32;
  const int d0 = blockIdx.y * 32;
  const int bh = blockIdx.z;
  const int b = bh >> 4, h = bh & 15;
  const int tx = threadIdx.x, ty = threadIdx.y;
#pragma unroll
  for (int i = 0; i < 4; ++i)
    tile[ty + 8 * i][tx] =
        qkv[(size_t)(b * 2048 + n0 + ty + 8 * i) * 3072 + 2048 + h * 64 + d0 + tx];
  __syncthreads();
#pragma unroll
  for (int i = 0; i < 4; ++i)
    vt[(size_t)(bh * 64 + d0 + ty + 8 * i) * 2048 + n0 + tx] = tile[tx][ty + 8 * i];
}

// ---------------------------------------------------------------------------
// Flash attention, WMMA bf16. Block = 128 thr = 4 waves; each wave owns 16
// query rows for one (b,h). Key chunks of 32: S = Q K^T (4 wmma), online
// softmax (shfl_xor row reductions over the C layout), P routed through LDS
// (C-frag -> A-frag relayout), O += P V (4 wmma against V^T).
// Grid: (2048/64, B*H).
// ---------------------------------------------------------------------------
__global__ __launch_bounds__(128) void flash_attn_kernel(
    const bf16_t* __restrict__ qkv, const bf16_t* __restrict__ vt,
    bf16_t* __restrict__ out)
{
  __shared__ __align__(16) bf16_t pbuf[4][16 * 32];
  const int lane = threadIdx.x & 31;
  const int wave = threadIdx.x >> 5;
  const int half = lane >> 4;
  const int r    = lane & 15;
  const int bh   = blockIdx.y;
  const int b = bh >> 4, h = bh & 15;
  const int q0 = blockIdx.x * 64 + wave * 16;

  const bf16_t* qp = qkv + (size_t)(b * 2048 + q0 + r) * 3072 + h * 64;
  const v16bf aq0 = ld_a_frag(qp + half * 8);        // d 0..31
  const v16bf aq1 = ld_a_frag(qp + 32 + half * 8);   // d 32..63

  const bf16_t* kbase = qkv + (size_t)(b * 2048) * 3072 + 1024 + h * 64;
  const bf16_t* vbase = vt + (size_t)bh * 64 * 2048;

  v8f o[4] = {};
  float mrow[8], lrow[8];
#pragma unroll
  for (int i = 0; i < 8; ++i) { mrow[i] = -3.0e38f; lrow[i] = 0.0f; }

  const float SC = 0.125f;  // DIM_HEAD^-0.5
  bf16_t* pl = pbuf[wave];

  for (int jc = 0; jc < 2048; jc += 32) {
    const bf16_t* kr0 = kbase + (size_t)(jc + r) * 3072 + half * 8;
    const bf16_t* kr1 = kr0 + (size_t)16 * 3072;
    const v16bf kb00 = ld_b_frag(kr0);        // keys jc..+15,  d 0..31
    const v16bf kb01 = ld_b_frag(kr0 + 32);   //                d 32..63
    const v16bf kb10 = ld_b_frag(kr1);        // keys jc+16..+31
    const v16bf kb11 = ld_b_frag(kr1 + 32);
    v8f s0 = {}, s1 = {};
    s0 = wmma_bf16(aq0, kb00, s0);
    s0 = wmma_bf16(aq1, kb01, s0);
    s1 = wmma_bf16(aq0, kb10, s1);
    s1 = wmma_bf16(aq1, kb11, s1);

    float al[8];
#pragma unroll
    for (int i = 0; i < 8; ++i) {
      const float x0 = s0[i] * SC;
      const float x1 = s1[i] * SC;
      float mx = fmaxf(x0, x1);
#pragma unroll
      for (int msk = 1; msk < 16; msk <<= 1)
        mx = fmaxf(mx, __shfl_xor(mx, msk, 32));
      const float mn = fmaxf(mrow[i], mx);
      al[i] = __expf(mrow[i] - mn);
      mrow[i] = mn;
      const float p0 = __expf(x0 - mn);
      const float p1 = __expf(x1 - mn);
      s0[i] = p0; s1[i] = p1;
      float rs = p0 + p1;
#pragma unroll
      for (int msk = 1; msk < 16; msk <<= 1)
        rs += __shfl_xor(rs, msk, 32);
      lrow[i] = lrow[i] * al[i] + rs;
    }
#pragma unroll
    for (int f = 0; f < 4; ++f)
#pragma unroll
      for (int i = 0; i < 8; ++i)
        o[f][i] *= al[i];

    __syncthreads();                 // prior reads of pl done (cross-lane)
#pragma unroll
    for (int i = 0; i < 8; ++i) {
      const int prow = half * 8 + i;
      pl[prow * 32 + r]      = (bf16_t)s0[i];
      pl[prow * 32 + 16 + r] = (bf16_t)s1[i];
    }
    __syncthreads();                 // P tile visible to all lanes
    const v16bf pf = ld_a_frag(pl + r * 32 + half * 8);
#pragma unroll
    for (int f = 0; f < 4; ++f) {
      const v16bf vb =
          ld_b_frag(vbase + (size_t)(f * 16 + r) * 2048 + jc + half * 8);
      o[f] = wmma_bf16(pf, vb, o[f]);
    }
  }

  float inv[8];
#pragma unroll
  for (int i = 0; i < 8; ++i) inv[i] = 1.0f / lrow[i];
#pragma unroll
  for (int f = 0; f < 4; ++f) {
#pragma unroll
    for (int t = 0; t < 8; ++t) {
      const int q = q0 + half * 8 + t;
      const int d = f * 16 + r;
      out[(size_t)(b * 2048 + q) * 1024 + h * 64 + d] = (bf16_t)(o[f][t] * inv[t]);
    }
  }
}

// ---------------------------------------------------------------------------
// Host: full 8-layer forward pass.
// ---------------------------------------------------------------------------
extern "C" void kernel_launch(void* const* d_in, const int* in_sizes, int n_in,
                              void* d_out, int out_size, void* d_ws, size_t ws_size,
                              hipStream_t stream) {
  (void)in_sizes; (void)n_in; (void)out_size; (void)ws_size;
  const float* x    = (const float*)d_in[0];
  const float* Wqkv = (const float*)d_in[1];
  const float* Wout = (const float*)d_in[2];
  const float* bout = (const float*)d_in[3];
  const float* g1   = (const float*)d_in[4];
  const float* b1n  = (const float*)d_in[5];
  const float* g2   = (const float*)d_in[6];
  const float* b2n  = (const float*)d_in[7];
  const float* W1   = (const float*)d_in[8];
  const float* bb1  = (const float*)d_in[9];
  const float* W2   = (const float*)d_in[10];
  const float* bb2  = (const float*)d_in[11];

  char* ws = (char*)d_ws;
  float*  xb   = (float*) (ws);                       // 16 MB residual stream f32
  bf16_t* hbuf = (bf16_t*)(ws + (16ull << 20));       //  8 MB LN output bf16
  bf16_t* wt   = (bf16_t*)(ws + (24ull << 20));       //  8 MB W^T bf16 scratch
  bf16_t* big  = (bf16_t*)(ws + (32ull << 20));       // 32 MB qkv / mlp-hidden bf16
  bf16_t* vt   = (bf16_t*)(ws + (64ull << 20));       //  8 MB V^T bf16
  bf16_t* attn = (bf16_t*)(ws + (72ull << 20));       //  8 MB attn out bf16

  hipMemcpyAsync(xb, x, (size_t)4096 * 1024 * 4, hipMemcpyDeviceToDevice, stream);

  const dim3 tb(32, 8);
  for (int L = 0; L < 8; ++L) {
    // ---- attention ----
    layernorm_kernel<<<4096, 256, 0, stream>>>(xb, g1 + L * 1024, b1n + L * 1024, hbuf);
    transpose_convert_kernel<<<dim3(3072 / 32, 1024 / 32), tb, 0, stream>>>(
        Wqkv + (size_t)L * 1024 * 3072, wt, 1024, 3072);
    gemm_bf16_kernel<0><<<dim3(3072 / 128, 4096 / 128), 256, 0, stream>>>(
        hbuf, wt, nullptr, nullptr, big, 4096, 3072, 1024);
    build_vt_kernel<<<dim3(64, 2, 32), tb, 0, stream>>>(big, vt);
    flash_attn_kernel<<<dim3(32, 32), 128, 0, stream>>>(big, vt, attn);
    transpose_convert_kernel<<<dim3(1024 / 32, 1024 / 32), tb, 0, stream>>>(
        Wout + (size_t)L * 1024 * 1024, wt, 1024, 1024);
    gemm_bf16_kernel<2><<<dim3(1024 / 128, 4096 / 128), 256, 0, stream>>>(
        attn, wt, bout + L * 1024, xb, xb, 4096, 1024, 1024);
    // ---- MLP ----
    layernorm_kernel<<<4096, 256, 0, stream>>>(xb, g2 + L * 1024, b2n + L * 1024, hbuf);
    transpose_convert_kernel<<<dim3(4096 / 32, 1024 / 32), tb, 0, stream>>>(
        W1 + (size_t)L * 1024 * 4096, wt, 1024, 4096);
    gemm_bf16_kernel<1><<<dim3(4096 / 128, 4096 / 128), 256, 0, stream>>>(
        hbuf, wt, bb1 + L * 4096, nullptr, big, 4096, 4096, 1024);
    transpose_convert_kernel<<<dim3(1024 / 32, 4096 / 32), tb, 0, stream>>>(
        W2 + (size_t)L * 4096 * 1024, wt, 4096, 1024);
    gemm_bf16_kernel<2><<<dim3(1024 / 128, 4096 / 128), 256, 0, stream>>>(
        big, wt, bb2 + L * 1024, xb, xb, 4096, 1024, 4096);
  }
  hipMemcpyAsync(d_out, xb, (size_t)4096 * 1024 * 4, hipMemcpyDeviceToDevice, stream);
}
